// LocalTransformerEncoderLayer_10471130268444
// MI455X (gfx1250) — matrix operand order repaired and checked by reading, not compiled, not run on hardware
//
#include <hip/hip_runtime.h>
#include <hip/hip_bf16.h>
#include <stdint.h>

using bf16 = __bf16;
typedef __attribute__((ext_vector_type(16))) __bf16 v16bf;
typedef __attribute__((ext_vector_type(8)))  float  v8f;
typedef __attribute__((ext_vector_type(4)))  unsigned int v4u;
typedef __attribute__((ext_vector_type(8)))  unsigned int v8u;

#define D_MODEL 1024
#define SEQ     4096
#define BATCH   4
#define NHEADS  16
#define HDIM    64
#define WINDOW  512
#define MTOT    (BATCH * SEQ)   // 16384 rows
#define DFF     (4 * D_MODEL)   // 4096

// ---------------------------------------------------------------------------
// WMMA helper: D = A(16x32 bf16) x B(32x16 bf16) + C(16x16 f32)
// ---------------------------------------------------------------------------
__device__ __forceinline__ v8f wmma_bf16(v16bf a, v16bf b, v8f c) {
  return __builtin_amdgcn_wmma_f32_16x16x32_bf16(
      /*neg_a=*/false, a, /*neg_b=*/false, b,
      /*c_mod=*/(short)0, c, /*reuse_a=*/false, /*reuse_b=*/false);
}

// A-matrix 16x32 bf16 fragment (ISA 7.12.2): lane half 0 holds K {0..7,16..23},
// half 1 holds K {8..15,24..31}.  p = row base (+ k offset), row = lane&15.
__device__ __forceinline__ v16bf frag_ld_a(const bf16* p, int hf) {
  union { v16bf v; uint4 q[2]; } u;
  u.q[0] = *reinterpret_cast<const uint4*>(p + (hf << 3));
  u.q[1] = *reinterpret_cast<const uint4*>(p + 16 + (hf << 3));
  return u.v;
}

// B-matrix 32x16 bf16 fragment: lane half 0 holds K {0..7,16..23},
// half 1 holds K {8..15,24..31}.  p = column base (row-major [N][K] storage).
__device__ __forceinline__ v16bf frag_ld_b(const bf16* p, int hf) {
  union { v16bf v; uint4 q[2]; } u;
  const bf16* q = p + (hf << 3);
  u.q[0] = *reinterpret_cast<const uint4*>(q);
  u.q[1] = *reinterpret_cast<const uint4*>(q + 16);
  return u.v;
}

// ---------------------------------------------------------------------------
// Tensor Data Mover: DMA a 2D tile (tile1 rows x tile0 elems, bf16) from
// global memory into LDS, inserting 4 DWORDs of pad every 32 DWORDs so a
// 64-elem row lands with a 72-elem (144B) LDS stride (bank-conflict free,
// matches frag_ld_* indexing).  D# layout per CDNA5 ISA sec 8.3/8.4.
// ---------------------------------------------------------------------------
__device__ __forceinline__ void tdm_load_tile(unsigned lds_off,
                                              const bf16* gaddr,
                                              unsigned tensor_d0,
                                              unsigned tensor_d1,
                                              unsigned tile0, unsigned tile1,
                                              unsigned stride0) {
  unsigned long long ga = (unsigned long long)(uintptr_t)gaddr;
  v4u g0;
  g0[0] = 1u;                                   // count=1 (valid), rest 0
  g0[1] = lds_off;                              // LDS byte address
  g0[2] = (unsigned)(ga & 0xffffffffu);         // global_addr[31:0]
  g0[3] = (unsigned)((ga >> 32) & 0x01ffffffu)  // global_addr[56:32]
        | (2u << 30);                           // type=2 ("image")
  v8u g1;
  g1[0] = (1u << 16)                            // data_size=1 -> 2 bytes
        | (1u << 20)                            // pad_enable
        | (4u << 22)                            // pad_interval: 32 DWORDs
        | (3u << 25);                           // pad_amount:   4 DWORDs
  g1[1] = (tensor_d0 & 0xffffu) << 16;          // tensor_dim0[15:0]
  g1[2] = (tensor_d0 >> 16) | ((tensor_d1 & 0xffffu) << 16);
  g1[3] = (tensor_d1 >> 16) | (tile0 << 16);    // tile_dim0
  g1[4] = tile1;                                // tile_dim1 (tile_dim2=0)
  g1[5] = stride0;                              // tensor_dim0_stride[31:0]
  g1[6] = 0u;                                   // stride0[47:32], dim1_stride lo
  g1[7] = 0u;
  asm volatile("tensor_load_to_lds %0, %1" ::"s"(g0), "s"(g1) : "memory");
}

// ---------------------------------------------------------------------------
// fp32 -> bf16 cast
// ---------------------------------------------------------------------------
__global__ __launch_bounds__(256) void cast_bf16_kernel(
    const float* __restrict__ in, bf16* __restrict__ out, int n) {
  int i = (blockIdx.x * 256 + threadIdx.x) * 4;
  if (i >= n) return;
  float4 f = *reinterpret_cast<const float4*>(in + i);
  out[i + 0] = (bf16)f.x;
  out[i + 1] = (bf16)f.y;
  out[i + 2] = (bf16)f.z;
  out[i + 3] = (bf16)f.w;
}

// ---------------------------------------------------------------------------
// Tiled WMMA GEMM: C[M,N] = A[M,K](bf16) * Bw[N,K]^T(bf16) + bias, opt. ReLU
// Block 256 threads = 8 waves; tile 128x128x64; wave tile 64x32 (4x2 accs).
// Tiles are DMAed into double-buffered LDS by the TDM (wave 0 issues),
// overlapping tile t+1's DMA with tile t's WMMA work.
// ---------------------------------------------------------------------------
constexpr int GBM = 128, GBN = 128, GBK = 64;
constexpr int LDT = GBK + 8;  // 72 elems = 144B padded stride (set by TDM pad)
constexpr size_t GEMM_LDS_BYTES = (size_t)2 * (GBM + GBN) * LDT * sizeof(bf16);

template <bool RELU, bool OUTBF>
__global__ __launch_bounds__(256) void gemm_kernel(
    const bf16* __restrict__ A, const bf16* __restrict__ Bw,
    const float* __restrict__ bias, void* __restrict__ Cout,
    int M, int N, int K) {
  extern __shared__ char smem_raw[];
  bf16* smem = reinterpret_cast<bf16*>(smem_raw);
  bf16* Asb[2] = {smem, smem + GBM * LDT};
  bf16* Bsb[2] = {smem + 2 * GBM * LDT, smem + 2 * GBM * LDT + GBN * LDT};

  const int tid  = threadIdx.x;
  const int lane = tid & 31;
  const int wv   = tid >> 5;
  const int wm   = wv >> 2;  // 0..1 (M)
  const int wn   = wv & 3;   // 0..3 (N)
  const int hf   = lane >> 4;
  const int l16  = lane & 15;
  const int rowC = blockIdx.y * GBM;
  const int colC = blockIdx.x * GBN;
  const bf16* Abase = A + (size_t)rowC * K;
  const bf16* Bbase = Bw + (size_t)colC * K;

  v8f zacc = {};
  v8f acc[4][2];
#pragma unroll
  for (int i = 0; i < 4; ++i)
#pragma unroll
    for (int j = 0; j < 2; ++j) acc[i][j] = zacc;

  const int ktiles = K / GBK;
  if (wv == 0) {
    tdm_load_tile((unsigned)(uintptr_t)Asb[0], Abase, (unsigned)K, (unsigned)M,
                  GBK, GBM, (unsigned)K);
    tdm_load_tile((unsigned)(uintptr_t)Bsb[0], Bbase, (unsigned)K, (unsigned)N,
                  GBK, GBN, (unsigned)K);
  }
  int cur = 0;
  for (int t = 0; t < ktiles; ++t) {
    if (wv == 0) {
      if (t + 1 < ktiles) {
        tdm_load_tile((unsigned)(uintptr_t)Asb[cur ^ 1],
                      Abase + (size_t)(t + 1) * GBK, (unsigned)K, (unsigned)M,
                      GBK, GBM, (unsigned)K);
        tdm_load_tile((unsigned)(uintptr_t)Bsb[cur ^ 1],
                      Bbase + (size_t)(t + 1) * GBK, (unsigned)K, (unsigned)N,
                      GBK, GBN, (unsigned)K);
        __builtin_amdgcn_s_wait_tensorcnt((short)2);  // tile t complete
      } else {
        __builtin_amdgcn_s_wait_tensorcnt((short)0);
      }
    }
    __syncthreads();  // publish LDS tile t to all waves
    const bf16* Acur = Asb[cur];
    const bf16* Bcur = Bsb[cur];
#pragma unroll
    for (int kk = 0; kk < GBK; kk += 32) {
      v16bf af[4], bfg[2];
#pragma unroll
      for (int mi = 0; mi < 4; ++mi)
        af[mi] = frag_ld_a(&Acur[(wm * 64 + mi * 16 + l16) * LDT + kk], hf);
#pragma unroll
      for (int ni = 0; ni < 2; ++ni)
        bfg[ni] = frag_ld_b(&Bcur[(wn * 32 + ni * 16 + l16) * LDT + kk], hf);
#pragma unroll
      for (int mi = 0; mi < 4; ++mi)
#pragma unroll
        for (int ni = 0; ni < 2; ++ni)
          acc[mi][ni] = wmma_bf16(af[mi], bfg[ni], acc[mi][ni]);
    }
    __syncthreads();  // all reads of tile t done; buffer reusable
    cur ^= 1;
  }

#pragma unroll
  for (int mi = 0; mi < 4; ++mi) {
#pragma unroll
    for (int ni = 0; ni < 2; ++ni) {
      int col  = colC + wn * 32 + ni * 16 + l16;
      float bv = bias ? bias[col] : 0.f;
#pragma unroll
      for (int r = 0; r < 8; ++r) {
        int row = rowC + wm * 64 + mi * 16 + r + hf * 8;
        float v = acc[mi][ni][r] + bv;
        if (RELU) v = fmaxf(v, 0.f);
        if (OUTBF)
          reinterpret_cast<bf16*>(Cout)[(size_t)row * N + col] = (bf16)v;
        else
          reinterpret_cast<float*>(Cout)[(size_t)row * N + col] = v;
      }
    }
  }
}

// ---------------------------------------------------------------------------
// Local windowed attention (causal, look_backward=1), flash-style.
// Grid: (qblock 0..3, window 0..7, b*16+h 0..63); 256 thr = 8 waves;
// wave handles 16 q rows, loops kv in 128-row LDS chunks / 32-row WMMA steps.
// K chunks arrive via TDM (padded to 72-elem stride); V is transposed
// manually into LDS so PV B-fragments are contiguous reads.
// ---------------------------------------------------------------------------
constexpr int LDK2 = HDIM + 8;   // 72  (Ks: [kv][d], stride set by TDM pad)
constexpr int LDVT = 128 + 8;    // 136 (Vt: [d][kv], transposed)
constexpr int LDP  = 32 + 8;     // 40  (per-wave P staging 16x32)

__global__ __launch_bounds__(256) void attn_kernel(
    const bf16* __restrict__ Qb, const bf16* __restrict__ Kb,
    const bf16* __restrict__ Vb, float* __restrict__ Ob) {
  __shared__ __align__(16) bf16 Ks[128 * LDK2];
  __shared__ __align__(16) bf16 Vt[HDIM * LDVT];
  __shared__ __align__(16) bf16 Ps[8 * 16 * LDP];
  const int tid  = threadIdx.x;
  const int lane = tid & 31;
  const int wv   = tid >> 5;
  const int hf   = lane >> 4, l16 = lane & 15;
  const int qb = blockIdx.x;      // 128-row q block within window
  const int w  = blockIdx.y;      // window index
  const int bh = blockIdx.z;
  const int b = bh >> 4, h = bh & 15;
  const size_t rowBase = (size_t)b * SEQ;
  const int qLoc0 = qb * 128 + wv * 16;   // q offset within window
  const int qRow0 = w * WINDOW + qLoc0;

  // Q fragments straight from global (rows across lanes, d contiguous)
  const bf16* qp = Qb + (rowBase + qRow0 + l16) * D_MODEL + h * HDIM;
  v16bf qf0 = frag_ld_a(qp, hf);
  v16bf qf1 = frag_ld_a(qp + 32, hf);

  v8f zacc = {};
  float mrow[8], lrow[8];
  v8f o[4];
#pragma unroll
  for (int r = 0; r < 8; ++r) { mrow[r] = -1e30f; lrow[r] = 0.f; }
#pragma unroll
  for (int ni = 0; ni < 4; ++ni) o[ni] = zacc;

  const int c0   = (w == 0) ? 4 : 0;   // window 0 has no previous window
  const int cmax = qb + 5;             // causal upper bound in 128-chunks
  for (int c = c0; c < cmax; ++c) {
    __syncthreads();  // previous chunk's LDS reads complete
    const int sBase = (w - 1) * WINDOW + c * 128;  // absolute kv row
    if (wv == 0) {
      const bf16* kp0 = Kb + (rowBase + sBase) * D_MODEL + h * HDIM;
      tdm_load_tile((unsigned)(uintptr_t)&Ks[0], kp0, D_MODEL,
                    (unsigned)MTOT, HDIM, 128, D_MODEL);
    }
    // V transposed into LDS by the vector pipe (TDM cannot transpose)
#pragma unroll
    for (int it = 0; it < 4; ++it) {
      int o_ = tid + it * 256;
      int r  = o_ >> 3;
      int c8 = (o_ & 7) << 3;
      const bf16* vp = Vb + (rowBase + sBase + r) * D_MODEL + h * HDIM + c8;
      union { uint4 u; bf16 e[8]; } vu;
      vu.u = *reinterpret_cast<const uint4*>(vp);
#pragma unroll
      for (int e = 0; e < 8; ++e) Vt[(c8 + e) * LDVT + r] = vu.e[e];
    }
    if (wv == 0) __builtin_amdgcn_s_wait_tensorcnt((short)0);
    __syncthreads();

    for (int sc = 0; sc < 4; ++sc) {
      const int jbase = c * 128 + sc * 32;  // local kv pos in [0,1024)
      // S = (Q * scale) K^T  -> two 16x16 f32 tiles
      v8f s0 = zacc, s1 = zacc;
      {
        const bf16* kb0 = &Ks[(sc * 32 + l16) * LDK2];
        s0 = wmma_bf16(qf0, frag_ld_b(kb0, hf), s0);
        s0 = wmma_bf16(qf1, frag_ld_b(kb0 + 32, hf), s0);
        const bf16* kb1 = &Ks[(sc * 32 + 16 + l16) * LDK2];
        s1 = wmma_bf16(qf0, frag_ld_b(kb1, hf), s1);
        s1 = wmma_bf16(qf1, frag_ld_b(kb1 + 32, hf), s1);
      }
      const float scale = 0.125f;  // 64^-0.5
      const int qcmp = WINDOW + qLoc0;  // valid iff j <= qcmp + rowoff
#pragma unroll
      for (int r = 0; r < 8; ++r) {
        int qlim = qcmp + r + hf * 8;
        float v0 = s0[r] * scale;
        float v1 = s1[r] * scale;
        if (jbase + l16 > qlim)      v0 = -1e30f;
        if (jbase + 16 + l16 > qlim) v1 = -1e30f;
        float tm = fmaxf(v0, v1);
#pragma unroll
        for (int msk = 1; msk < 16; msk <<= 1)
          tm = fmaxf(tm, __shfl_xor(tm, msk, 32));
        float mnew  = fmaxf(mrow[r], tm);
        float alpha = __expf(mrow[r] - mnew);
        mrow[r] = mnew;
        float p0 = __expf(v0 - mnew);
        float p1 = __expf(v1 - mnew);
        float ps = p0 + p1;
#pragma unroll
        for (int msk = 1; msk < 16; msk <<= 1) ps += __shfl_xor(ps, msk, 32);
        lrow[r] = lrow[r] * alpha + ps;
        s0[r] = p0; s1[r] = p1;
#pragma unroll
        for (int ni = 0; ni < 4; ++ni) o[ni][r] *= alpha;
      }
      // C-layout P -> LDS -> A-layout fragment (in-wave, DS in order)
      bf16* pw = &Ps[wv * 16 * LDP];
#pragma unroll
      for (int r = 0; r < 8; ++r) {
        pw[(r + 8 * hf) * LDP + l16]      = (bf16)s0[r];
        pw[(r + 8 * hf) * LDP + 16 + l16] = (bf16)s1[r];
      }
      asm volatile("s_wait_dscnt 0" ::: "memory");
      v16bf pf = frag_ld_a(&pw[l16 * LDP], hf);
#pragma unroll
      for (int ni = 0; ni < 4; ++ni) {
        v16bf vf = frag_ld_b(&Vt[(ni * 16 + l16) * LDVT + sc * 32], hf);
        o[ni] = wmma_bf16(pf, vf, o[ni]);
      }
    }
  }

#pragma unroll
  for (int r = 0; r < 8; ++r) {
    float inv  = 1.f / lrow[r];
    size_t row = rowBase + qRow0 + r + 8 * hf;
#pragma unroll
    for (int ni = 0; ni < 4; ++ni)
      Ob[row * D_MODEL + h * HDIM + ni * 16 + l16] = o[ni][r] * inv;
  }
}

// ---------------------------------------------------------------------------
// LayerNorm over D_MODEL=1024: x = mulX*X + (X2? X2 : 0); outputs f32 / bf16
// ---------------------------------------------------------------------------
__global__ __launch_bounds__(256) void ln_kernel(
    const float* __restrict__ X, const float* __restrict__ X2,
    const float* __restrict__ g, const float* __restrict__ be,
    float* __restrict__ Yf, bf16* __restrict__ Yb, float mulX) {
  __shared__ float rs[8], rss[8];
  const int row  = blockIdx.x;
  const int tid  = threadIdx.x;
  const int lane = tid & 31, wv = tid >> 5;
  const float* xp = X + (size_t)row * D_MODEL;
  float vals[4];
  float s = 0.f, ss = 0.f;
#pragma unroll
  for (int i = 0; i < 4; ++i) {
    int c   = tid + i * 256;
    float v = mulX * xp[c];
    if (X2) v += X2[(size_t)row * D_MODEL + c];
    vals[i] = v;
    s += v;
    ss += v * v;
  }
#pragma unroll
  for (int m = 1; m < 32; m <<= 1) {
    s  += __shfl_xor(s, m, 32);
    ss += __shfl_xor(ss, m, 32);
  }
  if (lane == 0) { rs[wv] = s; rss[wv] = ss; }
  __syncthreads();
  float st = 0.f, sst = 0.f;
#pragma unroll
  for (int i = 0; i < 8; ++i) { st += rs[i]; sst += rss[i]; }
  float mu   = st * (1.0f / D_MODEL);
  float var  = sst * (1.0f / D_MODEL) - mu * mu;
  float rstd = rsqrtf(var + 1e-5f);
#pragma unroll
  for (int i = 0; i < 4; ++i) {
    int c   = tid + i * 256;
    float y = (vals[i] - mu) * rstd * g[c] + be[c];
    if (Yf) Yf[(size_t)row * D_MODEL + c] = y;
    if (Yb) Yb[(size_t)row * D_MODEL + c] = (bf16)y;
  }
}

// ---------------------------------------------------------------------------
// Host-side launch
// ---------------------------------------------------------------------------
extern "C" void kernel_launch(void* const* d_in, const int* in_sizes, int n_in,
                              void* d_out, int out_size, void* d_ws,
                              size_t ws_size, hipStream_t stream) {
  (void)in_sizes; (void)n_in; (void)out_size; (void)ws_size;
  const float* src = (const float*)d_in[0];
  const float* wq  = (const float*)d_in[1];
  const float* bq  = (const float*)d_in[2];
  const float* wk  = (const float*)d_in[3];
  const float* bk  = (const float*)d_in[4];
  const float* wvp = (const float*)d_in[5];
  const float* bv  = (const float*)d_in[6];
  const float* w1  = (const float*)d_in[7];
  const float* b1  = (const float*)d_in[8];
  const float* w2  = (const float*)d_in[9];
  const float* b2  = (const float*)d_in[10];
  const float* g1  = (const float*)d_in[11];
  const float* be1 = (const float*)d_in[12];
  const float* g2  = (const float*)d_in[13];
  const float* be2 = (const float*)d_in[14];
  float* outp = (float*)d_out;

  char* p = (char*)d_ws;
  auto alloc = [&](size_t bytes) {
    char* r = p;
    p += (bytes + 255) & ~size_t(255);
    return r;
  };
  bf16* srcb = (bf16*)alloc((size_t)MTOT * D_MODEL * 2);
  bf16* qbuf = (bf16*)alloc((size_t)MTOT * D_MODEL * 2);
  bf16* kbuf = (bf16*)alloc((size_t)MTOT * D_MODEL * 2);
  bf16* vbuf = (bf16*)alloc((size_t)MTOT * D_MODEL * 2);
  bf16* xbf  = (bf16*)alloc((size_t)MTOT * D_MODEL * 2);
  bf16* hbuf = (bf16*)alloc((size_t)MTOT * DFF * 2);
  bf16* wqb  = (bf16*)alloc((size_t)D_MODEL * D_MODEL * 2);
  bf16* wkb  = (bf16*)alloc((size_t)D_MODEL * D_MODEL * 2);
  bf16* wvb  = (bf16*)alloc((size_t)D_MODEL * D_MODEL * 2);
  bf16* w1b  = (bf16*)alloc((size_t)DFF * D_MODEL * 2);
  bf16* w2b  = (bf16*)alloc((size_t)D_MODEL * DFF * 2);
  float* attnb = (float*)alloc((size_t)MTOT * D_MODEL * 4);
  float* xbuf  = (float*)alloc((size_t)MTOT * D_MODEL * 4);
  float* ffnb  = (float*)alloc((size_t)MTOT * D_MODEL * 4);

  auto cast = [&](const float* in, bf16* out, size_t n) {
    cast_bf16_kernel<<<dim3((unsigned)(n / 1024)), dim3(256), 0, stream>>>(
        in, out, (int)n);
  };
  cast(src, srcb, (size_t)MTOT * D_MODEL);
  cast(wq, wqb, (size_t)D_MODEL * D_MODEL);
  cast(wk, wkb, (size_t)D_MODEL * D_MODEL);
  cast(wvp, wvb, (size_t)D_MODEL * D_MODEL);
  cast(w1, w1b, (size_t)DFF * D_MODEL);
  cast(w2, w2b, (size_t)D_MODEL * DFF);

  dim3 blk(256);
  dim3 gQKV(D_MODEL / GBN, MTOT / GBM);  // (8,128)
  gemm_kernel<false, true><<<gQKV, blk, GEMM_LDS_BYTES, stream>>>(
      srcb, wqb, bq, qbuf, MTOT, D_MODEL, D_MODEL);
  gemm_kernel<false, true><<<gQKV, blk, GEMM_LDS_BYTES, stream>>>(
      srcb, wkb, bk, kbuf, MTOT, D_MODEL, D_MODEL);
  gemm_kernel<false, true><<<gQKV, blk, GEMM_LDS_BYTES, stream>>>(
      srcb, wvb, bv, vbuf, MTOT, D_MODEL, D_MODEL);

  attn_kernel<<<dim3(4, 8, BATCH * NHEADS), blk, 0, stream>>>(qbuf, kbuf, vbuf,
                                                              attnb);

  ln_kernel<<<dim3(MTOT), blk, 0, stream>>>(attnb, nullptr, g1, be1, xbuf, xbf,
                                            2.0f);

  gemm_kernel<true, true><<<dim3(DFF / GBN, MTOT / GBM), blk, GEMM_LDS_BYTES,
                            stream>>>(xbf, w1b, b1, hbuf, MTOT, DFF, D_MODEL);
  gemm_kernel<false, false><<<dim3(D_MODEL / GBN, MTOT / GBM), blk,
                              GEMM_LDS_BYTES, stream>>>(hbuf, w2b, b2, ffnb,
                                                        MTOT, D_MODEL, DFF);

  ln_kernel<<<dim3(MTOT), blk, 0, stream>>>(xbuf, ffnb, g2, be2, outp, nullptr,
                                            1.0f);
}
